// CompGCNClassifier_49675591746336
// MI455X (gfx1250) — matrix operand order repaired and checked by reading, compiled, not run on hardware
//
#include <hip/hip_runtime.h>
#include <hip/hip_bf16.h>

typedef __attribute__((ext_vector_type(16))) __bf16 v16bf;
typedef __attribute__((ext_vector_type(8)))  __bf16 v8bf;
typedef __attribute__((ext_vector_type(8)))  float  v8f;

union AFrag { v16bf v; __bf16 e[16]; };

#define WMMA_BF16(a, b, c) \
  __builtin_amdgcn_wmma_f32_16x16x32_bf16(false, (a), false, (b), (short)0, (c), false, false)

__device__ __forceinline__ float gelu_exact(float x) {
  return 0.5f * x * (1.0f + erff(x * 0.70710678118654752f));
}
__device__ __forceinline__ float sigm(float x) { return 1.0f / (1.0f + expf(-x)); }

// Load 8 contiguous fp32, convert to bf16 (two aligned float4 loads).
__device__ __forceinline__ void load_run_f32(const float* __restrict__ p, __bf16* d) {
  float4 a = ((const float4*)p)[0];
  float4 b = ((const float4*)p)[1];
  d[0] = (__bf16)a.x; d[1] = (__bf16)a.y; d[2] = (__bf16)a.z; d[3] = (__bf16)a.w;
  d[4] = (__bf16)b.x; d[5] = (__bf16)b.y; d[6] = (__bf16)b.z; d[7] = (__bf16)b.w;
}

// Load 8 contiguous bf16 (one aligned 16B load). Works for global and LDS ptrs.
__device__ __forceinline__ void load_run_bf16(const __bf16* p, __bf16* d) {
  v8bf x = *(const v8bf*)p;
#pragma unroll
  for (int i = 0; i < 8; ++i) d[i] = x[i];
}

// ---- small utility kernels -------------------------------------------------
__global__ void cvt_f32_bf16_k(const float* __restrict__ s, __bf16* __restrict__ d, int n) {
  int i = blockIdx.x * blockDim.x + threadIdx.x;
  if (i < n) d[i] = (__bf16)s[i];
}
__global__ void zero_f32_k(float* __restrict__ p, int n) {
  int i = blockIdx.x * blockDim.x + threadIdx.x;
  if (i < n) p[i] = 0.0f;
}

// ---- kernel 1: edge message GEMM + GELU + scatter-add ----------------------
// msg = gelu([nf[src]|nf[tgt]|rel_emb[lab]] @ msg_W^T + b);  agg[tgt] += msg
__global__ void msg_kernel(const float* __restrict__ nf,
                           const int* __restrict__ src, const int* __restrict__ tgt,
                           const int* __restrict__ lab, const float* __restrict__ rel,
                           const __bf16* __restrict__ Wbf, const float* __restrict__ bias,
                           float* __restrict__ agg, int ntiles) {
  int wave = blockIdx.x * (blockDim.x >> 5) + (threadIdx.x >> 5);
  if (wave >= ntiles) return;
  int lane = threadIdx.x & 31;
  int m    = lane & 15;
  int hi   = lane >> 4;
  int kbA  = hi * 16;   // A-frag K sub-offset
  int kbB  = hi * 16;   // B-frag K sub-offset
  int e0   = wave * 16;
  int e    = e0 + m;

  // warm WGP$ with the first weight rows this lane will stream
  __builtin_prefetch(Wbf + (size_t)m * 384, 0, 0);
  __builtin_prefetch(Wbf + (size_t)(64 + m) * 384, 0, 0);

  const float* pS = nf  + (size_t)src[e] * 128;
  const float* pT = nf  + (size_t)tgt[e] * 128;
  const float* pR = rel + (size_t)lab[e] * 128;

  auto loadA = [&](int kt, AFrag& a) {
    int k1 = kt * 32 + kbA;
    const float* p1 = (k1 < 128) ? (pS + k1) : (k1 < 256) ? (pT + k1 - 128) : (pR + k1 - 256);
    int k2 = k1 + 8;
    const float* p2 = (k2 < 128) ? (pS + k2) : (k2 < 256) ? (pT + k2 - 128) : (pR + k2 - 256);
    load_run_f32(p1, &a.e[0]);
    load_run_f32(p2, &a.e[8]);
  };

  v8f acc[8] = {};
  AFrag a_cur, a_nxt;
  loadA(0, a_cur);
#pragma unroll
  for (int kt = 0; kt < 12; ++kt) {
    if (kt < 11) loadA(kt + 1, a_nxt);          // overlap gather+cvt with WMMA
    v16bf bfr[8];
#pragma unroll
    for (int n0 = 0; n0 < 8; ++n0)
      bfr[n0] = *(const v16bf*)(Wbf + (size_t)(n0 * 16 + m) * 384 + kt * 32 + kbB);
#pragma unroll
    for (int n0 = 0; n0 < 8; ++n0)
      acc[n0] = WMMA_BF16(a_cur.v, bfr[n0], acc[n0]);
    a_cur = a_nxt;
  }

#pragma unroll
  for (int n0 = 0; n0 < 8; ++n0) {
    int   n  = n0 * 16 + m;
    float bs = bias[n];
#pragma unroll
    for (int r = 0; r < 8; ++r) {
      int   row = hi * 8 + r;                 // edge within tile
      int   te  = tgt[e0 + row];
      float g   = gelu_exact(acc[n0][r] + bs);
      unsafeAtomicAdd(&agg[(size_t)te * 128 + n], g);
    }
  }
}

// ---- kernel 2: GRU cell over nodes -----------------------------------------
__global__ void gru_kernel(const float* __restrict__ agg, const float* __restrict__ nf,
                           const __bf16* __restrict__ Wih, const __bf16* __restrict__ Whh,
                           const float* __restrict__ bih, const float* __restrict__ bhh,
                           __bf16* __restrict__ nfu, int ntiles) {
  int wave = blockIdx.x * (blockDim.x >> 5) + (threadIdx.x >> 5);
  if (wave >= ntiles) return;
  int lane = threadIdx.x & 31;
  int m    = lane & 15;
  int hi   = lane >> 4;
  int kbA  = hi * 16;
  int kbB  = hi * 16;
  int n0d  = wave * 16;

  __builtin_prefetch(Wih + (size_t)m * 128, 0, 0);
  __builtin_prefetch(Whh + (size_t)m * 128, 0, 0);

  const float* pA = agg + (size_t)(n0d + m) * 128;
  const float* pH = nf  + (size_t)(n0d + m) * 128;

  AFrag aA[4], aH[4];
#pragma unroll
  for (int kt = 0; kt < 4; ++kt) {
    load_run_f32(pA + kt * 32 + kbA,      &aA[kt].e[0]);
    load_run_f32(pA + kt * 32 + kbA + 8,  &aA[kt].e[8]);
    load_run_f32(pH + kt * 32 + kbA,      &aH[kt].e[0]);
    load_run_f32(pH + kt * 32 + kbA + 8,  &aH[kt].e[8]);
  }

#pragma unroll 2
  for (int c0 = 0; c0 < 8; ++c0) {
    v8f ir = {}, iz = {}, in_ = {}, hr = {}, hz = {}, hn = {};
    int crow = c0 * 16 + m;
#pragma unroll
    for (int kt = 0; kt < 4; ++kt) {
      int koff = kt * 32 + kbB;
      v16bf b0  = *(const v16bf*)(Wih + (size_t)(crow)       * 128 + koff);
      v16bf b1  = *(const v16bf*)(Wih + (size_t)(crow + 128) * 128 + koff);
      v16bf b2  = *(const v16bf*)(Wih + (size_t)(crow + 256) * 128 + koff);
      v16bf c0v = *(const v16bf*)(Whh + (size_t)(crow)       * 128 + koff);
      v16bf c1v = *(const v16bf*)(Whh + (size_t)(crow + 128) * 128 + koff);
      v16bf c2v = *(const v16bf*)(Whh + (size_t)(crow + 256) * 128 + koff);
      ir  = WMMA_BF16(aA[kt].v, b0,  ir);
      iz  = WMMA_BF16(aA[kt].v, b1,  iz);
      in_ = WMMA_BF16(aA[kt].v, b2,  in_);
      hr  = WMMA_BF16(aH[kt].v, c0v, hr);
      hz  = WMMA_BF16(aH[kt].v, c1v, hz);
      hn  = WMMA_BF16(aH[kt].v, c2v, hn);
    }
    int   c   = c0 * 16 + m;
    float bir = bih[c], biz = bih[c + 128], bin = bih[c + 256];
    float bhr = bhh[c], bhz = bhh[c + 128], bhn = bhh[c + 256];
#pragma unroll
    for (int r = 0; r < 8; ++r) {
      int   node = n0d + hi * 8 + r;
      float rg = sigm(ir[r] + bir + hr[r] + bhr);
      float zg = sigm(iz[r] + biz + hz[r] + bhz);
      float ng = tanhf(in_[r] + bin + rg * (hn[r] + bhn));
      float hp = nf[(size_t)node * 128 + c];
      float hv = (1.0f - zg) * ng + zg * hp;
      nfu[(size_t)node * 128 + c] = (__bf16)hv;
    }
  }
}

// ---- kernel 3: classifier (GEMM1 + GELU -> LDS -> GEMM2) -------------------
#define LDSTR 136  // padded half-stride per hc row (spreads LDS banks)
__global__ void cls_kernel(const __bf16* __restrict__ nfu, const float* __restrict__ ef,
                           const int* __restrict__ src, const int* __restrict__ tgt,
                           const __bf16* __restrict__ W1, const float* __restrict__ b1,
                           const __bf16* __restrict__ W2, const float* __restrict__ b2,
                           float* __restrict__ out, int ntiles) {
  __shared__ __bf16 lds[2][16 * LDSTR];
  int wave = blockIdx.x * (blockDim.x >> 5) + (threadIdx.x >> 5);
  if (wave >= ntiles) return;
  int slot = (threadIdx.x >> 5) & 1;
  int lane = threadIdx.x & 31;
  int m    = lane & 15;
  int hi   = lane >> 4;
  int kbA  = hi * 16;
  int kbB  = hi * 16;
  int e0   = wave * 16;
  int e    = e0 + m;

  __builtin_prefetch(W1 + (size_t)m * 384, 0, 0);
  __builtin_prefetch(W2 + (size_t)m * 128, 0, 0);

  const __bf16* pS = nfu + (size_t)src[e] * 128;
  const __bf16* pT = nfu + (size_t)tgt[e] * 128;
  const float*  pE = ef  + (size_t)e * 128;

  auto loadA = [&](int kt, AFrag& a) {
    int k1 = kt * 32 + kbA;
    if (k1 < 128)      load_run_bf16(pS + k1,       &a.e[0]);
    else if (k1 < 256) load_run_bf16(pT + k1 - 128, &a.e[0]);
    else               load_run_f32 (pE + k1 - 256, &a.e[0]);
    int k2 = k1 + 8;
    if (k2 < 128)      load_run_bf16(pS + k2,       &a.e[8]);
    else if (k2 < 256) load_run_bf16(pT + k2 - 128, &a.e[8]);
    else               load_run_f32 (pE + k2 - 256, &a.e[8]);
  };

  v8f acc[8] = {};
  AFrag a_cur, a_nxt;
  loadA(0, a_cur);
#pragma unroll
  for (int kt = 0; kt < 12; ++kt) {
    if (kt < 11) loadA(kt + 1, a_nxt);
    v16bf bfr[8];
#pragma unroll
    for (int n0 = 0; n0 < 8; ++n0)
      bfr[n0] = *(const v16bf*)(W1 + (size_t)(n0 * 16 + m) * 384 + kt * 32 + kbB);
#pragma unroll
    for (int n0 = 0; n0 < 8; ++n0)
      acc[n0] = WMMA_BF16(a_cur.v, bfr[n0], acc[n0]);
    a_cur = a_nxt;
  }

  // GELU + stage hc tile (16 x 128 bf16) into LDS in row-major form.
  __bf16* tl = &lds[slot][0];
#pragma unroll
  for (int n0 = 0; n0 < 8; ++n0) {
    int   n  = n0 * 16 + m;
    float bs = b1[n];
#pragma unroll
    for (int r = 0; r < 8; ++r) {
      float g = gelu_exact(acc[n0][r] + bs);
      tl[(hi * 8 + r) * LDSTR + n] = (__bf16)g;  // same-wave RAW; compiler inserts s_wait_dscnt
    }
  }

  // GEMM2: [16x128] x [128x16]
  v8f o = {};
  const __bf16* arow = tl + m * LDSTR;
  v16bf b2f[4];
#pragma unroll
  for (int kt = 0; kt < 4; ++kt)
    b2f[kt] = *(const v16bf*)(W2 + (size_t)m * 128 + kt * 32 + kbB);
#pragma unroll
  for (int kt = 0; kt < 4; ++kt) {
    AFrag a;
    load_run_bf16(arow + kt * 32 + kbA,      &a.e[0]);
    load_run_bf16(arow + kt * 32 + kbA + 8,  &a.e[8]);
    o = WMMA_BF16(a.v, b2f[kt], o);
  }
  float bb = b2[m];
#pragma unroll
  for (int r = 0; r < 8; ++r)
    out[(size_t)(e0 + hi * 8 + r) * 16 + m] = o[r] + bb;
}

// ---- launcher --------------------------------------------------------------
extern "C" void kernel_launch(void* const* d_in, const int* in_sizes, int n_in,
                              void* d_out, int out_size, void* d_ws, size_t ws_size,
                              hipStream_t stream) {
  const float* nf   = (const float*)d_in[0];
  const float* ef   = (const float*)d_in[1];
  const int*   ei   = (const int*)  d_in[2];
  const int*   lab  = (const int*)  d_in[3];
  const float* rel  = (const float*)d_in[4];
  const float* msgW = (const float*)d_in[5];
  const float* msgb = (const float*)d_in[6];
  const float* wih  = (const float*)d_in[7];
  const float* whh  = (const float*)d_in[8];
  const float* bih  = (const float*)d_in[9];
  const float* bhh  = (const float*)d_in[10];
  const float* W1   = (const float*)d_in[11];
  const float* b1   = (const float*)d_in[12];
  const float* W2   = (const float*)d_in[13];
  const float* b2   = (const float*)d_in[14];
  float* out = (float*)d_out;

  const int N = in_sizes[0] / 128;
  const int E = in_sizes[1] / 128;
  const int* src = ei;
  const int* tgt = ei + E;

  char* ws = (char*)d_ws;
  size_t o = 0;
  auto carve = [&](size_t bytes) { char* p = ws + o; o = (o + bytes + 255) & ~(size_t)255; return p; };
  float*  agg    = (float*) carve((size_t)N * 128 * 4);
  __bf16* nfu    = (__bf16*)carve((size_t)N * 128 * 2);
  __bf16* msgWb  = (__bf16*)carve(128 * 384 * 2);
  __bf16* W1b    = (__bf16*)carve(128 * 384 * 2);
  __bf16* W2b    = (__bf16*)carve(16 * 128 * 2);
  __bf16* wihb   = (__bf16*)carve(384 * 128 * 2);
  __bf16* whhb   = (__bf16*)carve(384 * 128 * 2);

  // 1) weight conversion f32 -> bf16
  cvt_f32_bf16_k<<<(128 * 384 + 255) / 256, 256, 0, stream>>>(msgW, msgWb, 128 * 384);
  cvt_f32_bf16_k<<<(128 * 384 + 255) / 256, 256, 0, stream>>>(W1,   W1b,   128 * 384);
  cvt_f32_bf16_k<<<(16 * 128 + 255)  / 256, 256, 0, stream>>>(W2,   W2b,   16 * 128);
  cvt_f32_bf16_k<<<(384 * 128 + 255) / 256, 256, 0, stream>>>(wih,  wihb,  384 * 128);
  cvt_f32_bf16_k<<<(384 * 128 + 255) / 256, 256, 0, stream>>>(whh,  whhb,  384 * 128);

  // 2) zero agg
  zero_f32_k<<<((size_t)N * 128 + 255) / 256, 256, 0, stream>>>(agg, N * 128);

  // 3) message + scatter
  int etiles = E / 16;
  msg_kernel<<<(etiles + 1) / 2, 64, 0, stream>>>(nf, src, tgt, lab, rel, msgWb, msgb, agg, etiles);

  // 4) GRU update
  int ntilesN = N / 16;
  gru_kernel<<<(ntilesN + 1) / 2, 64, 0, stream>>>(agg, nf, wihb, whhb, bih, bhh, nfu, ntilesN);

  // 5) classifier
  cls_kernel<<<(etiles + 1) / 2, 64, 0, stream>>>(nfu, ef, src, tgt, W1b, b1, W2b, b2, out, etiles);
}
